// RandomAttention_56573309223861
// MI455X (gfx1250) — matrix operand order repaired and checked by reading, not compiled, not run on hardware
//
#include <hip/hip_runtime.h>
#include <hip/hip_bf16.h>

// RandomAttention for MI455X (gfx1250, wave32).
// out[b,h,s,:] = (sum_r softmax_r(q[b,h,s]·k[b,h,idx[s,r]]/sqrt(D))) * v[b,h,s,:]
//
// One wave32 per query row. Scores via v_wmma_f32_16x16x32_bf16:
//   A = 16 gathered key rows (16x32 bf16 chunk), B = q replicated over 16 cols,
//   two accumulators cover keys 0-15 / 16-31, 4 K-chunks cover D=128.
// Softmax over 32 keys uses the C-layout key partition across lane halves:
// per-thread 16-element reduce + one shfl_xor(16). Output pass is pure
// bandwidth (v read NT, out written NT).

typedef __attribute__((ext_vector_type(16))) __bf16 v16bf;
typedef __attribute__((ext_vector_type(8)))  float  v8f;
typedef __attribute__((ext_vector_type(4)))  float  v4f;   // clang vector: legal for NT builtins

#define B_ 2
#define H_ 16
#define S_ 2048
#define D_ 128
#define R_ 32
#define NQ_ (B_ * H_ * S_)   // 65536 query rows

__device__ __forceinline__ v16bf pack16_bf16(v4f a, v4f b, v4f c, v4f d) {
  v16bf r;
  r[0]  = (__bf16)a.x; r[1]  = (__bf16)a.y; r[2]  = (__bf16)a.z; r[3]  = (__bf16)a.w;
  r[4]  = (__bf16)b.x; r[5]  = (__bf16)b.y; r[6]  = (__bf16)b.z; r[7]  = (__bf16)b.w;
  r[8]  = (__bf16)c.x; r[9]  = (__bf16)c.y; r[10] = (__bf16)c.z; r[11] = (__bf16)c.w;
  r[12] = (__bf16)d.x; r[13] = (__bf16)d.y; r[14] = (__bf16)d.z; r[15] = (__bf16)d.w;
  return r;
}

__global__ __launch_bounds__(256) void rand_attn_wmma_kernel(
    const float* __restrict__ q, const float* __restrict__ k,
    const float* __restrict__ v, const int* __restrict__ ridx,
    float* __restrict__ out) {
  const int lane = threadIdx.x & 31;
  const int qid  = (int)(blockIdx.x * (blockDim.x >> 5)) + (int)(threadIdx.x >> 5);
  const int s    = qid & (S_ - 1);   // S = 2^11
  const int bh   = qid >> 11;
  const int hi   = lane >> 4;        // lane half: 0 or 1
  const int m    = lane & 15;        // row within the 16-key A tile

  const float* __restrict__ qrow = q + (size_t)qid * D_;
  const float* __restrict__ vrow = v + (size_t)qid * D_;
  // Overlap the streaming v read with the gather/score math.
  __builtin_prefetch(vrow + lane * 4, 0, 0);   // -> global_prefetch_b8

  // Gather indices: key tile 0 = idx[s,0..15], tile 1 = idx[s,16..31].
  const int kid0 = ridx[s * R_ + m];
  const int kid1 = ridx[s * R_ + 16 + m];
  const float* __restrict__ kr0 = k + ((size_t)bh * S_ + (size_t)kid0) * D_;
  const float* __restrict__ kr1 = k + ((size_t)bh * S_ + (size_t)kid1) * D_;

  v8f acc0 = {};   // scores for keys 0-15 (replicated over 16 N-columns)
  v8f acc1 = {};   // scores for keys 16-31
#pragma unroll
  for (int c = 0; c < 4; ++c) {
    // A-matrix (16x32 bf16): lane holds row M=lane%16; element e maps to
    // K = e + 8*hi + (e>=8 ? 8 : 0)  => runs [dA..dA+7] and [dA+16..dA+23].
    const int dA = 32 * c + 8 * hi;
    const v4f* p0 = (const v4f*)(kr0 + dA);
    const v4f* p1 = (const v4f*)(kr1 + dA);
    v16bf A0 = pack16_bf16(p0[0], p0[1], p0[4], p0[5]);
    v16bf A1 = pack16_bf16(p1[0], p1[1], p1[4], p1[5]);
    // B-matrix (32x16 bf16): column N=lane%16 (all columns = q); element e
    // maps to K = e + 8*hi + (e>=8 ? 8 : 0) => runs [dB..dB+7] and [dB+16..dB+23].
    const int dB = 32 * c + 8 * hi;
    const v4f* pq = (const v4f*)(qrow + dB);
    v16bf Bm = pack16_bf16(pq[0], pq[1], pq[4], pq[5]);

    acc0 = __builtin_amdgcn_wmma_f32_16x16x32_bf16(
        false, A0, false, Bm, (short)0, acc0, false, false);
    acc1 = __builtin_amdgcn_wmma_f32_16x16x32_bf16(
        false, A1, false, Bm, (short)0, acc1, false, false);
  }

  // C layout: thread t, element i => key m = i + 8*(t>=16) (acc0) / +16 (acc1).
  // Lane halves partition the 32 keys: threads 0-15 hold {0-7,16-23},
  // threads 16-31 hold {8-15,24-31}; each key replicated over 16 columns.
  const float scale = 0.08838834764831845f;  // 1/sqrt(128)
  float sc[16];
#pragma unroll
  for (int i = 0; i < 8; ++i) {
    sc[i]     = acc0[i] * scale;
    sc[8 + i] = acc1[i] * scale;
  }
  float smax = -3.402823466e38f;
#pragma unroll
  for (int i = 0; i < 16; ++i) smax = fmaxf(smax, sc[i]);
  smax = fmaxf(smax, __shfl_xor(smax, 16, 32));  // global max over 32 keys

  float esum_loc = 0.0f;
  float ev[16];
#pragma unroll
  for (int i = 0; i < 16; ++i) {
    ev[i] = __expf(sc[i] - smax);
    esum_loc += ev[i];
  }
  // Key sets of lane l and lane l^16 are disjoint -> pair-sum = full sum.
  const float esum = esum_loc + __shfl_xor(esum_loc, 16, 32);

  // Reference: probs = e/esum elementwise, then sum_r probs  (≈ 1.0 in fp32).
  float psum_loc = 0.0f;
#pragma unroll
  for (int i = 0; i < 16; ++i) psum_loc += ev[i] / esum;
  const float psum = psum_loc + __shfl_xor(psum_loc, 16, 32);

  // out = psum * v  (v read once, out written once -> nontemporal).
  const v4f* vp = (const v4f*)vrow;
  v4f* op = (v4f*)(out + (size_t)qid * D_);
  v4f t = __builtin_nontemporal_load(vp + lane);
  t *= psum;
  __builtin_nontemporal_store(t, op + lane);
}

extern "C" void kernel_launch(void* const* d_in, const int* in_sizes, int n_in,
                              void* d_out, int out_size, void* d_ws, size_t ws_size,
                              hipStream_t stream) {
  const float* q    = (const float*)d_in[0];
  const float* k    = (const float*)d_in[1];
  const float* v    = (const float*)d_in[2];
  const int*   ridx = (const int*)d_in[3];
  float*       out  = (float*)d_out;

  // 1 wave per query row, 8 waves (256 threads) per block.
  const int waves_per_block = 8;
  dim3 block(32 * waves_per_block);
  dim3 grid(NQ_ / waves_per_block);  // 8192 blocks, all full -> EXEC all-1s
  rand_attn_wmma_kernel<<<grid, block, 0, stream>>>(q, k, v, ridx, out);
}
